// MultiHeadAttention_16647293239356
// MI455X (gfx1250) — matrix operand order repaired and checked by reading, not compiled, not run on hardware
//
#include <hip/hip_runtime.h>
#include <hip/hip_bf16.h>

// ---------------------------------------------------------------------------
// MHA forward for MI455X (gfx1250): bf16 WMMA (v_wmma_f32_16x16x32_bf16) for
// all matmuls, fp32 softmax (flash-style streaming), fp32 output.
// Ping-pong (2x unrolled) software pipeline: no register-rotation copies.
// B=2, L=2048, HIDDEN=1024, HEADS=16, HEAD=64.
// ---------------------------------------------------------------------------

typedef __bf16 bf16_t;
typedef __attribute__((ext_vector_type(16))) __bf16 v16bf;
typedef __attribute__((ext_vector_type(8)))  __bf16 v8bf;
typedef __attribute__((ext_vector_type(4)))  __bf16 v4bf;
typedef __attribute__((ext_vector_type(8)))  float  v8f;

#define L_SEQ   2048
#define HID     1024
#define NHEAD   16
#define DHEAD   64
#define NROWS   4096   // B*L

__device__ __forceinline__ v16bf concat8(v8bf lo, v8bf hi) {
  return __builtin_shufflevector(lo, hi, 0,1,2,3,4,5,6,7,8,9,10,11,12,13,14,15);
}

// A operand (16x32, MxK), row-major bf16 source with leading dim `ld` (elems).
// ISA layout: lane l (0..15) -> row M=l, K in {0..7}u{16..23};
//             lane l (16..31)-> row M=l-16, K in {8..15}u{24..31}.
__device__ __forceinline__ v16bf frag_a(const bf16_t* __restrict__ base, int ld) {
  const int l  = threadIdx.x & 31;
  const int m  = l & 15;
  const int kb = (l >> 4) * 8;
  const bf16_t* p = base + m * ld + kb;
  v8bf lo = *(const v8bf*)(p);        // K = kb .. kb+7
  v8bf hi = *(const v8bf*)(p + 16);   // K = kb+16 .. kb+23
  return concat8(lo, hi);
}

// B operand (32x16, KxN) for computing X @ Y^T where Y is row-major bf16 with
// K contiguous (leading dim `ld`). B[k][n] = Y[n][k].
// ISA layout: lane l -> column N=l&15; K in {kb..kb+7}u{kb+16..kb+23}, kb=8*(l>>4).
__device__ __forceinline__ v16bf frag_b(const bf16_t* __restrict__ base, int ld) {
  const int l  = threadIdx.x & 31;
  const int n  = l & 15;
  const int kb = (l >> 4) * 8;
  const bf16_t* p = base + n * ld + kb;
  v8bf lo = *(const v8bf*)(p);
  v8bf hi = *(const v8bf*)(p + 16);
  return concat8(lo, hi);
}

__device__ __forceinline__ v8f wmma_bf16(v16bf a, v16bf b, v8f c) {
  return __builtin_amdgcn_wmma_f32_16x16x32_bf16(false, a, false, b,
                                                 (short)0, c, false, false);
}

// ---------------------------------------------------------------------------
// Kernel 0: fp32 -> bf16 conversion (vectorized x4)
// ---------------------------------------------------------------------------
__global__ void __launch_bounds__(256) cvt_bf16_kernel(
    const float* __restrict__ in, bf16_t* __restrict__ out, int n4) {
  int i = blockIdx.x * 256 + threadIdx.x;
  if (i < n4) {
    float4 f = ((const float4*)in)[i];
    v4bf o;
    o[0] = (bf16_t)f.x; o[1] = (bf16_t)f.y; o[2] = (bf16_t)f.z; o[3] = (bf16_t)f.w;
    ((v4bf*)out)[i] = o;
  }
}

// ---------------------------------------------------------------------------
// Kernel 1: fused QKV projection.
// One wave computes a 32x64 tile of q/k/v = x @ W^T + b. Ping-pong operand
// sets A/B over the k-loop (2 chunks per iteration, no rotation copies).
// z = blockIdx.z selects matrix: 0->Q (scaled by 1/8, [B,H,L,64]),
// 1->K ([B,H,L,64]), 2->V (stored transposed [B,H,64,L]).
// ---------------------------------------------------------------------------
__global__ void __launch_bounds__(32) qkv_kernel(
    const bf16_t* __restrict__ xbf,                       // [4096,1024]
    const bf16_t* __restrict__ wqb, const bf16_t* __restrict__ wkb,
    const bf16_t* __restrict__ wvb,
    const float*  __restrict__ bq,  const float* __restrict__ bk,
    const float*  __restrict__ bv,
    bf16_t* __restrict__ Qs, bf16_t* __restrict__ Kb, bf16_t* __restrict__ Vt) {
  const int mt = blockIdx.x;       // 0..127  (32 rows each)
  const int ns = blockIdx.y;       // 0..15   (64 cols each)
  const int z  = blockIdx.z;       // 0..2

  const bf16_t* w    = (z == 0) ? wqb : (z == 1) ? wkb : wvb;
  const float*  bvec = (z == 0) ? bq  : (z == 1) ? bk  : bv;

  v8f acc[2][4];
#pragma unroll
  for (int mi = 0; mi < 2; ++mi)
#pragma unroll
    for (int t = 0; t < 4; ++t)
#pragma unroll
      for (int r = 0; r < 8; ++r) acc[mi][t][r] = 0.0f;

  const bf16_t* arow0 = xbf + (size_t)mt * 32 * HID;
  const bf16_t* arow1 = arow0 + 16 * HID;
  const bf16_t* wrow  = w + (size_t)(ns * 64) * HID;

  // pipeline stage: prefetch k=knext into (ya0,ya1,yb), compute with (xa0,xa1,xb)
  auto gemm_step = [&](v16bf& xa0, v16bf& xa1, v16bf* xb,
                       v16bf& ya0, v16bf& ya1, v16bf* yb, int knext) {
    ya0 = frag_a(arow0 + knext, HID);
    ya1 = frag_a(arow1 + knext, HID);
#pragma unroll
    for (int t = 0; t < 4; ++t)
      yb[t] = frag_b(wrow + (size_t)t * 16 * HID + knext, HID);
#pragma unroll
    for (int t = 0; t < 4; ++t) {
      acc[0][t] = wmma_bf16(xa0, xb[t], acc[0][t]);
      acc[1][t] = wmma_bf16(xa1, xb[t], acc[1][t]);
    }
  };

  // prologue: fetch k-chunk 0 into set A
  v16bf aA0 = frag_a(arow0, HID);
  v16bf aA1 = frag_a(arow1, HID);
  v16bf bA[4];
#pragma unroll
  for (int t = 0; t < 4; ++t) bA[t] = frag_b(wrow + (size_t)t * 16 * HID, HID);
  v16bf aB0, aB1, bB[4];

  for (int k0 = 0; k0 < HID; k0 += 64) {
    gemm_step(aA0, aA1, bA, aB0, aB1, bB, k0 + 32);          // use A, fill B
    const int k2 = (k0 + 64 < HID) ? (k0 + 64) : 0;          // clamped (dead last)
    gemm_step(aB0, aB1, bB, aA0, aA1, bA, k2);               // use B, fill A
  }

  const int lane = threadIdx.x & 31;
  const int col  = lane & 15;
  const int hs   = lane >> 4;
  float bias_t[4];
#pragma unroll
  for (int t = 0; t < 4; ++t) bias_t[t] = bvec[ns * 64 + t * 16 + col];

#pragma unroll
  for (int mi = 0; mi < 2; ++mi) {
#pragma unroll
    for (int t = 0; t < 4; ++t) {
#pragma unroll
      for (int r = 0; r < 8; ++r) {
        const int n = ns * 64 + t * 16 + col;         // hidden col
        const int m = mt * 32 + mi * 16 + r + 8 * hs; // flat (b,l) row
        float val = acc[mi][t][r] + bias_t[t];
        if (z == 0) val *= 0.125f;                    // HEAD^-0.5 folded into Q
        const int bb = m >> 11, ll = m & (L_SEQ - 1);
        const int hh = n >> 6,  dd = n & 63;
        if (z == 2) {
          Vt[(((size_t)bb * NHEAD + hh) * DHEAD + dd) * L_SEQ + ll] = (bf16_t)val;
        } else {
          bf16_t* dst = (z == 0) ? Qs : Kb;
          dst[(((size_t)bb * NHEAD + hh) * L_SEQ + ll) * DHEAD + dd] = (bf16_t)val;
        }
      }
    }
  }
}

// ---------------------------------------------------------------------------
// Kernel 2: flash attention. One wave owns a 16-query tile of one (b,h);
// streams keys/values 32 at a time with running-max softmax.
// Ping-pong K-fragment sets across kv blocks (2 blocks per loop iteration);
// V fragments + bias issued early so softmax overlaps the fetches.
// ---------------------------------------------------------------------------
__global__ void __launch_bounds__(128) attn_kernel(
    const bf16_t* __restrict__ Qs, const bf16_t* __restrict__ Kb,
    const bf16_t* __restrict__ Vt, const float* __restrict__ bias,
    bf16_t* __restrict__ Obf) {
  __shared__ float ldsP[4][16 * 32];   // per-wave P repack buffer (8 KB)

  const int wave = threadIdx.x >> 5;
  const int lane = threadIdx.x & 31;
  const int col  = lane & 15;
  const int hs   = lane >> 4;

  const int bh = blockIdx.x;                 // 0..31 = b*16+h
  const int qt = blockIdx.y * 4 + wave;      // 0..127
  const int b  = bh >> 4, h = bh & 15;
  const int q0 = qt * 16;

  const bf16_t* Qh = Qs + (size_t)bh * L_SEQ * DHEAD;
  const bf16_t* Kh = Kb + (size_t)bh * L_SEQ * DHEAD;
  const bf16_t* Vh = Vt + (size_t)bh * DHEAD * L_SEQ;

  // Q tile (16x64) as two A fragments, resident for the whole loop.
  const v16bf qa0 = frag_a(Qh + (size_t)q0 * DHEAD, DHEAD);
  const v16bf qa1 = frag_a(Qh + (size_t)q0 * DHEAD + 32, DHEAD);

  float m_r[8], l_r[8];
  v8f acc[4];
#pragma unroll
  for (int r = 0; r < 8; ++r) { m_r[r] = -1e30f; l_r[r] = 0.0f; }
#pragma unroll
  for (int t = 0; t < 4; ++t)
#pragma unroll
    for (int r = 0; r < 8; ++r) acc[t][r] = 0.0f;

  float* lp = ldsP[wave];

  // one kv block: use kfc (keys kv0..kv0+31), prefetch keys kvn into kfn_
  auto attn_step = [&](int kv0, int kvn, v16bf* kfc, v16bf* kfn_) {
    // ---- issue V fragments + bias loads for this block up front
    v16bf vb[4];
#pragma unroll
    for (int t = 0; t < 4; ++t)
      vb[t] = frag_b(Vh + (size_t)(t * 16) * L_SEQ + kv0, L_SEQ);
    float bvals0[8], bvals1[8];
#pragma unroll
    for (int r = 0; r < 8; ++r) {
      const int qq = q0 + r + 8 * hs;
      const float* bp = bias + (size_t)qq * L_SEQ + kv0 + col;
      bvals0[r] = bp[0];
      bvals1[r] = bp[16];
    }

    // ---- S = Q @ K^T : two 16x16 tiles (keys kv0..+15 and kv0+16..+31)
    v8f S0, S1;
#pragma unroll
    for (int r = 0; r < 8; ++r) { S0[r] = 0.0f; S1[r] = 0.0f; }
    S0 = wmma_bf16(qa0, kfc[0], S0);
    S0 = wmma_bf16(qa1, kfc[1], S0);
    S1 = wmma_bf16(qa0, kfc[2], S1);
    S1 = wmma_bf16(qa1, kfc[3], S1);

    // ---- prefetch next block's K fragments (into the other set)
#pragma unroll
    for (int j = 0; j < 2; ++j) {
      kfn_[2 * j]     = frag_b(Kh + (size_t)(kvn + 16 * j) * DHEAD, DHEAD);
      kfn_[2 * j + 1] = frag_b(Kh + (size_t)(kvn + 16 * j) * DHEAD + 32, DHEAD);
    }

    // ---- add attention bias (input [1,1,L,L], fp32)
#pragma unroll
    for (int r = 0; r < 8; ++r) { S0[r] += bvals0[r]; S1[r] += bvals1[r]; }

    // ---- streaming softmax update (row stats live on matching lane halves)
    float scale[8];
#pragma unroll
    for (int r = 0; r < 8; ++r) {
      float v = fmaxf(S0[r], S1[r]);
      v = fmaxf(v, __shfl_xor(v, 1, 32));
      v = fmaxf(v, __shfl_xor(v, 2, 32));
      v = fmaxf(v, __shfl_xor(v, 4, 32));
      v = fmaxf(v, __shfl_xor(v, 8, 32));
      const float mnew = fmaxf(m_r[r], v);
      const float sc = __expf(m_r[r] - mnew);
      const float p0 = __expf(S0[r] - mnew);
      const float p1 = __expf(S1[r] - mnew);
      S0[r] = p0; S1[r] = p1;
      float s = p0 + p1;
      s += __shfl_xor(s, 1, 32);
      s += __shfl_xor(s, 2, 32);
      s += __shfl_xor(s, 4, 32);
      s += __shfl_xor(s, 8, 32);
      l_r[r] = l_r[r] * sc + s;
      m_r[r] = mnew;
      scale[r] = sc;
    }
#pragma unroll
    for (int t = 0; t < 4; ++t)
#pragma unroll
      for (int r = 0; r < 8; ++r) acc[t][r] *= scale[r];

    // ---- repack P (C layout -> A layout) through LDS, fp32 -> bf16
#pragma unroll
    for (int r = 0; r < 8; ++r) {
      const int row = r + 8 * hs;
      lp[row * 32 + col]      = S0[r];
      lp[row * 32 + 16 + col] = S1[r];
    }
    asm volatile("s_wait_dscnt 0x0" ::: "memory");  // cross-lane LDS RAW
    v16bf pa;
    {
      const int m  = lane & 15;
      const int kb = (lane >> 4) * 8;
      const float* p = lp + m * 32 + kb;
      float4 c0 = *(const float4*)(p);
      float4 c1 = *(const float4*)(p + 4);
      float4 c2 = *(const float4*)(p + 16);
      float4 c3 = *(const float4*)(p + 20);
      pa[0]  = (bf16_t)c0.x; pa[1]  = (bf16_t)c0.y;
      pa[2]  = (bf16_t)c0.z; pa[3]  = (bf16_t)c0.w;
      pa[4]  = (bf16_t)c1.x; pa[5]  = (bf16_t)c1.y;
      pa[6]  = (bf16_t)c1.z; pa[7]  = (bf16_t)c1.w;
      pa[8]  = (bf16_t)c2.x; pa[9]  = (bf16_t)c2.y;
      pa[10] = (bf16_t)c2.z; pa[11] = (bf16_t)c2.w;
      pa[12] = (bf16_t)c3.x; pa[13] = (bf16_t)c3.y;
      pa[14] = (bf16_t)c3.z; pa[15] = (bf16_t)c3.w;
    }

    // ---- acc += P @ V  (V loaded up front; overlapped with softmax)
#pragma unroll
    for (int t = 0; t < 4; ++t) acc[t] = wmma_bf16(pa, vb[t], acc[t]);
  };

  // prologue: K fragments for kv block 0 into set A
  v16bf kfA[4], kfB[4];
#pragma unroll
  for (int j = 0; j < 2; ++j) {
    kfA[2 * j]     = frag_b(Kh + (size_t)(16 * j) * DHEAD, DHEAD);
    kfA[2 * j + 1] = frag_b(Kh + (size_t)(16 * j) * DHEAD + 32, DHEAD);
  }

  for (int kv0 = 0; kv0 < L_SEQ; kv0 += 64) {
    attn_step(kv0, kv0 + 32, kfA, kfB);                       // use A, fill B
    const int k2 = (kv0 + 64 < L_SEQ) ? (kv0 + 64) : 0;       // clamped (dead last)
    attn_step(kv0 + 32, k2, kfB, kfA);                        // use B, fill A
  }

  // ---- normalize and store combined-head layout [B,L,H*64] (bf16)
#pragma unroll
  for (int t = 0; t < 4; ++t) {
#pragma unroll
    for (int r = 0; r < 8; ++r) {
      const int row = r + 8 * hs;
      const float o = acc[t][r] / l_r[r];
      const size_t addr =
          ((size_t)(b * L_SEQ + q0 + row)) * HID + h * DHEAD + t * 16 + col;
      Obf[addr] = (bf16_t)o;
    }
  }
}

// ---------------------------------------------------------------------------
// Kernel 3: output projection  out = Obf @ Wo^T + bo  (fp32 out), 32x64/wave,
// ping-pong pipelined like the QKV kernel.
// ---------------------------------------------------------------------------
__global__ void __launch_bounds__(32) oproj_kernel(
    const bf16_t* __restrict__ Obf, const bf16_t* __restrict__ wob,
    const float* __restrict__ bo, float* __restrict__ out) {
  const int mt = blockIdx.x;   // 0..127
  const int ns = blockIdx.y;   // 0..15

  v8f acc[2][4];
#pragma unroll
  for (int mi = 0; mi < 2; ++mi)
#pragma unroll
    for (int t = 0; t < 4; ++t)
#pragma unroll
      for (int r = 0; r < 8; ++r) acc[mi][t][r] = 0.0f;

  const bf16_t* arow0 = Obf + (size_t)mt * 32 * HID;
  const bf16_t* arow1 = arow0 + 16 * HID;
  const bf16_t* wrow  = wob + (size_t)(ns * 64) * HID;

  auto gemm_step = [&](v16bf& xa0, v16bf& xa1, v16bf* xb,
                       v16bf& ya0, v16bf& ya1, v16bf* yb, int knext) {
    ya0 = frag_a(arow0 + knext, HID);
    ya1 = frag_a(arow1 + knext, HID);
#pragma unroll
    for (int t = 0; t < 4; ++t)
      yb[t] = frag_b(wrow + (size_t)t * 16 * HID + knext, HID);
#pragma unroll
    for (int t = 0; t < 4; ++t) {
      acc[0][t] = wmma_bf16(xa0, xb[t], acc[0][t]);
      acc[1][t] = wmma_bf16(xa1, xb[t], acc[1][t]);
    }
  };

  v16bf aA0 = frag_a(arow0, HID);
  v16bf aA1 = frag_a(arow1, HID);
  v16bf bA[4];
#pragma unroll
  for (int t = 0; t < 4; ++t) bA[t] = frag_b(wrow + (size_t)t * 16 * HID, HID);
  v16bf aB0, aB1, bB[4];

  for (int k0 = 0; k0 < HID; k0 += 64) {
    gemm_step(aA0, aA1, bA, aB0, aB1, bB, k0 + 32);
    const int k2 = (k0 + 64 < HID) ? (k0 + 64) : 0;
    gemm_step(aB0, aB1, bB, aA0, aA1, bA, k2);
  }

  const int lane = threadIdx.x & 31;
  const int col  = lane & 15;
  const int hs   = lane >> 4;
  float bias_t[4];
#pragma unroll
  for (int t = 0; t < 4; ++t) bias_t[t] = bo[ns * 64 + t * 16 + col];

#pragma unroll
  for (int mi = 0; mi < 2; ++mi) {
#pragma unroll
    for (int t = 0; t < 4; ++t) {
#pragma unroll
      for (int r = 0; r < 8; ++r) {
        const int n = ns * 64 + t * 16 + col;
        const int m = mt * 32 + mi * 16 + r + 8 * hs;
        out[(size_t)m * HID + n] = acc[mi][t][r] + bias_t[t];
      }
    }
  }
}

// ---------------------------------------------------------------------------
// Host launcher
// ---------------------------------------------------------------------------
extern "C" void kernel_launch(void* const* d_in, const int* in_sizes, int n_in,
                              void* d_out, int out_size, void* d_ws, size_t ws_size,
                              hipStream_t stream) {
  (void)in_sizes; (void)n_in; (void)out_size; (void)ws_size;
  const float* query = (const float*)d_in[0];   // [2,2048,1024]
  const float* bias  = (const float*)d_in[1];   // [1,1,2048,2048]
  const float* wq    = (const float*)d_in[2];
  const float* bq    = (const float*)d_in[3];
  const float* wk    = (const float*)d_in[4];
  const float* bk    = (const float*)d_in[5];
  const float* wv    = (const float*)d_in[6];
  const float* bv    = (const float*)d_in[7];
  const float* wo    = (const float*)d_in[8];
  const float* bo    = (const float*)d_in[9];
  float* out = (float*)d_out;

  // workspace carve-up (48 MB total)
  char* ws = (char*)d_ws;
  bf16_t* qbf = (bf16_t*)(ws);                          //  8 MB [4096,1024]
  bf16_t* wqb = (bf16_t*)(ws + (8u  << 20));            //  2 MB
  bf16_t* wkb = (bf16_t*)(ws + (10u << 20));            //  2 MB
  bf16_t* wvb = (bf16_t*)(ws + (12u << 20));            //  2 MB
  bf16_t* wob = (bf16_t*)(ws + (14u << 20));            //  2 MB
  bf16_t* Qs  = (bf16_t*)(ws + (16u << 20));            //  8 MB [B,H,L,64]
  bf16_t* Kbf = (bf16_t*)(ws + (24u << 20));            //  8 MB [B,H,L,64]
  bf16_t* Vt  = (bf16_t*)(ws + (32u << 20));            //  8 MB [B,H,64,L]
  bf16_t* Obf = (bf16_t*)(ws + (40u << 20));            //  8 MB [B,L,1024]

  // 0) fp32 -> bf16 conversions
  {
    const int nq4 = (NROWS * HID) / 4;     // 1,048,576
    const int nw4 = (HID * HID) / 4;       //   262,144
    cvt_bf16_kernel<<<(nq4 + 255) / 256, 256, 0, stream>>>(query, qbf, nq4);
    cvt_bf16_kernel<<<(nw4 + 255) / 256, 256, 0, stream>>>(wq, wqb, nw4);
    cvt_bf16_kernel<<<(nw4 + 255) / 256, 256, 0, stream>>>(wk, wkb, nw4);
    cvt_bf16_kernel<<<(nw4 + 255) / 256, 256, 0, stream>>>(wv, wvb, nw4);
    cvt_bf16_kernel<<<(nw4 + 255) / 256, 256, 0, stream>>>(wo, wob, nw4);
  }

  // 1) fused QKV projection: grid = (M tiles of 32, N strips of 64, {q,k,v})
  qkv_kernel<<<dim3(NROWS / 32, HID / 64, 3), 32, 0, stream>>>(
      qbf, wqb, wkb, wvb, bq, bk, bv, Qs, Kbf, Vt);

  // 2) flash attention: grid = (B*H, L/16/4), 4 waves per block
  attn_kernel<<<dim3(NHEAD * 2, L_SEQ / 16 / 4), 128, 0, stream>>>(
      Qs, Kbf, Vt, bias, Obf);

  // 3) output projection -> fp32 d_out
  oproj_kernel<<<dim3(NROWS / 32, HID / 64), 32, 0, stream>>>(
      Obf, wob, bo, out);
}
